// SelfAttention_30846455120707
// MI455X (gfx1250) — matrix operand order repaired and checked by reading, not compiled, not run on hardware
//
#include <hip/hip_runtime.h>

// ---------------------------------------------------------------------------
// Qwen3-style self-attention for gfx1250 (MI455X), f16 WMMA + fp32 accumulate
// ---------------------------------------------------------------------------

#define S_LEN 2048
#define HID   4096
#define NH    32
#define NKV   8
#define HD    128
#define EPS   1e-6f
#define LOG2_ROPE_BASE 19.931568569324174f   // log2(1e6)

typedef __attribute__((ext_vector_type(16))) _Float16 v16h;
typedef __attribute__((ext_vector_type(8)))  _Float16 v8h;
typedef __attribute__((ext_vector_type(8)))  float    v8f;

static __device__ __forceinline__ v16h ld2(const _Float16* p0, const _Float16* p1) {
    v8h a = *(const v8h*)p0;
    v8h b = *(const v8h*)p1;
    v16h r;
#pragma unroll
    for (int i = 0; i < 8; ++i) { r[i] = a[i]; r[i + 8] = b[i]; }
    return r;
}

static __device__ __forceinline__ v8f wmma16(v16h a, v16h b, v8f c) {
    return __builtin_amdgcn_wmma_f32_16x16x32_f16(
        /*neg_a=*/false, a, /*neg_b=*/false, b,
        /*c_mod=*/(short)0, c, /*reuse_a=*/false, /*reuse_b=*/false);
}

// ------------------------------- fp32 -> f16 --------------------------------
__global__ void cvt_f16(const float* __restrict__ in, _Float16* __restrict__ out, int n) {
    int stride = gridDim.x * blockDim.x;
    for (int i = blockIdx.x * blockDim.x + threadIdx.x; i < n; i += stride)
        out[i] = (_Float16)in[i];
}

// -------------------- generic WMMA GEMM: C = A * W^T ------------------------
// A: [M,K] f16 row-major.  W: [N,K] f16 row-major.  C: [M,N] f32 row-major.
// One wave computes a 32x64 tile: 2 A-tiles x 4 B-tiles, K stepped by 32.
// Indexed addressing (base + row*K + k0) keeps the loads on the GLOBAL
// encoding (pure LOADcnt) instead of FLAT (LOADcnt+DScnt).
__global__ __launch_bounds__(128) void gemm_wmma(const _Float16* __restrict__ A,
                                                 const _Float16* __restrict__ W,
                                                 float* __restrict__ C,
                                                 int M, int N, int K) {
    const int lane = threadIdx.x & 31;
    const int wid  = threadIdx.x >> 5;
    const int mt = M >> 5, nt = N >> 6;
    const int w = blockIdx.x * 4 + wid;
    if (w >= mt * nt) return;                     // wave-uniform
    const int m0 = (w % mt) << 5;
    const int n0 = (w / mt) << 6;
    const int li = lane & 15, hi = lane >> 4;

    const v8f zf = {0.f,0.f,0.f,0.f,0.f,0.f,0.f,0.f};
    v8f acc[2][4];
#pragma unroll
    for (int i = 0; i < 2; ++i)
#pragma unroll
        for (int j = 0; j < 4; ++j) acc[i][j] = zf;

    for (int k0 = 0; k0 < K; k0 += 32) {
        v16h a[2], b[4];
#pragma unroll
        for (int i = 0; i < 2; ++i) {
            const _Float16* p = A + (size_t)(m0 + i * 16 + li) * K + k0 + (hi ? 8 : 0);
            a[i] = ld2(p, p + 16);                // K=0..7/16..23 (lo half), +8 (hi)
        }
#pragma unroll
        for (int j = 0; j < 4; ++j) {
            const _Float16* p = W + (size_t)(n0 + j * 16 + li) * K + k0 + (hi ? 8 : 0);
            b[j] = ld2(p, p + 16);                // K=0..7/16..23 (lo half), +8 (hi)
            __builtin_prefetch(p + 128, 0, 0);    // global_prefetch, 4 K-steps ahead
        }
#pragma unroll
        for (int i = 0; i < 2; ++i)
#pragma unroll
            for (int j = 0; j < 4; ++j)
                acc[i][j] = wmma16(a[i], b[j], acc[i][j]);
    }

    // epilogue: one base pointer per lane, immediate offsets for the N-tiles
    float* cp = C + (size_t)(m0 + hi * 8) * N + n0 + li;
#pragma unroll
    for (int i = 0; i < 2; ++i) {
        float* cpi = cp + (size_t)(i * 16) * N;
#pragma unroll
        for (int r = 0; r < 8; ++r) {
            float* cpr = cpi + (size_t)r * N;
#pragma unroll
            for (int j = 0; j < 4; ++j)
                cpr[j * 16] = acc[i][j][r];
        }
    }
}

// ---------------- per-head RMS norm + RoPE, fp32 in -> f16 head-major -------
// in:  [S, nh*HD] fp32 ;  out: [nh][S][HD] f16.  One wave per (head,seq) row.
__global__ __launch_bounds__(256) void norm_rope(const float* __restrict__ in,
                                                 const float* __restrict__ nw,
                                                 _Float16* __restrict__ out, int nh) {
    const int lane = threadIdx.x & 31;
    const int wid  = threadIdx.x >> 5;
    const int row  = blockIdx.x * 8 + wid;
    if (row >= nh * S_LEN) return;
    const int h = row / S_LEN;
    const int s = row % S_LEN;

    float v[4];
#pragma unroll
    for (int i = 0; i < 4; ++i)
        v[i] = in[(size_t)s * (nh * HD) + h * HD + lane + 32 * i];

    float ss = v[0]*v[0] + v[1]*v[1] + v[2]*v[2] + v[3]*v[3];
#pragma unroll
    for (int off = 1; off < 32; off <<= 1) ss += __shfl_xor(ss, off, 32);
    const float rn = rsqrtf(ss * (1.0f / (float)HD) + EPS);
#pragma unroll
    for (int i = 0; i < 4; ++i)
        v[i] = v[i] * rn * nw[lane + 32 * i];

    float o[4];
#pragma unroll
    for (int i = 0; i < 4; ++i) {
        const int d = lane + 32 * i;
        const int j = d & 63;                                  // theta index
        const float th  = exp2f(-(float)j * (LOG2_ROPE_BASE / 64.0f));
        const float ang = (float)s * th;
        const float c = cosf(ang), sn = sinf(ang);
        o[i] = (i < 2) ? (v[i] * c - v[i + 2] * sn)
                       : (v[i] * c + v[i - 2] * sn);
    }
#pragma unroll
    for (int i = 0; i < 4; ++i)
        out[(size_t)h * S_LEN * HD + (size_t)s * HD + lane + 32 * i] = (_Float16)o[i];
}

// --------------------- V transpose: [S, KV*HD] f32 -> [KV][HD][S] f16 -------
__global__ void vtrans(const float* __restrict__ v32, _Float16* __restrict__ vT) {
    const int n = NKV * HD * S_LEN;
    int stride = gridDim.x * blockDim.x;
    for (int i = blockIdx.x * blockDim.x + threadIdx.x; i < n; i += stride) {
        const int t  = i & (S_LEN - 1);
        const int d  = (i >> 11) & (HD - 1);
        const int kv = i >> 18;
        vT[i] = (_Float16)v32[(size_t)t * (NKV * HD) + kv * HD + d];
    }
}

// ------------------------- flash causal GQA attention -----------------------
// qh: [H][S][D] f16, kh: [KV][S][D] f16, vT: [KV][D][S] f16, ctx: [S, H*D] f16
// One wave per (head, 16-query tile). Key blocks of 32.
__global__ __launch_bounds__(128) void attn(const _Float16* __restrict__ qh,
                                            const _Float16* __restrict__ kh,
                                            const _Float16* __restrict__ vT,
                                            _Float16* __restrict__ ctx) {
    __shared__ _Float16 plds[4][16 * 32];
    const int lane = threadIdx.x & 31;
    const int wid  = threadIdx.x >> 5;
    const int w  = blockIdx.x * 4 + wid;            // 0 .. H*(S/16)-1
    const int h  = w >> 7;                          // S/16 = 128 tiles/head
    const int q0 = (w & 127) << 4;
    const int kv = h >> 2;                          // GQA group of 4
    const int li = lane & 15, hi = lane >> 4;
    const float scale = 0.08838834764831845f;       // 1/sqrt(128)

    const _Float16* Q  = qh + (size_t)h  * S_LEN * HD;
    const _Float16* Km = kh + (size_t)kv * S_LEN * HD;
    const _Float16* Vm = vT + (size_t)kv * HD * S_LEN;
    _Float16* pw = &plds[wid][0];

    // preload the 16x128 Q tile as 4 WMMA A operands (K-chunks of 32)
    v16h aq[4];
#pragma unroll
    for (int kc = 0; kc < 4; ++kc) {
        const _Float16* p = Q + (size_t)(q0 + li) * HD + kc * 32 + (hi ? 8 : 0);
        aq[kc] = ld2(p, p + 16);
    }

    const v8f zf = {0.f,0.f,0.f,0.f,0.f,0.f,0.f,0.f};
    v8f acc[8];
#pragma unroll
    for (int t = 0; t < 8; ++t) acc[t] = zf;
    float mrow[8], lrow[8];
#pragma unroll
    for (int r = 0; r < 8; ++r) { mrow[r] = -3.0e38f; lrow[r] = 0.f; }

    const int tend = q0 + 16;                       // causal key count
    for (int t0 = 0; t0 < tend; t0 += 32) {
        // prefetch the next key block's rows for this lane
        __builtin_prefetch(Km + (size_t)(t0 + 32 + li) * HD, 0, 0);
        __builtin_prefetch(Km + (size_t)(t0 + 48 + li) * HD, 0, 0);

        v8f s0 = zf, s1 = zf;
#pragma unroll
        for (int kc = 0; kc < 4; ++kc) {
            const _Float16* b0 = Km + (size_t)(t0 + li)      * HD + kc * 32 + (hi ? 8 : 0);
            const _Float16* b1 = Km + (size_t)(t0 + 16 + li) * HD + kc * 32 + (hi ? 8 : 0);
            s0 = wmma16(aq[kc], ld2(b0, b0 + 16), s0);
            s1 = wmma16(aq[kc], ld2(b1, b1 + 16), s1);
        }
        // online softmax; row m = r + 8*hi lives in VGPR r across a 16-lane half
        float alpha[8];
#pragma unroll
        for (int r = 0; r < 8; ++r) {
            const int qi = q0 + r + hi * 8;
            float x0 = s0[r] * scale; if (t0 + li      > qi) x0 = -3.0e38f;
            float x1 = s1[r] * scale; if (t0 + 16 + li > qi) x1 = -3.0e38f;
            float mx = fmaxf(x0, x1);
#pragma unroll
            for (int off = 1; off < 16; off <<= 1) mx = fmaxf(mx, __shfl_xor(mx, off, 32));
            const float mn = fmaxf(mrow[r], mx);
            const float a  = __expf(mrow[r] - mn);
            const float p0 = __expf(x0 - mn);
            const float p1 = __expf(x1 - mn);
            float sum = p0 + p1;
#pragma unroll
            for (int off = 1; off < 16; off <<= 1) sum += __shfl_xor(sum, off, 32);
            lrow[r] = lrow[r] * a + sum;
            mrow[r] = mn;
            alpha[r] = a;
            const int m = r + hi * 8;
            pw[m * 32 + li]      = (_Float16)p0;    // stage P for transpose
            pw[m * 32 + 16 + li] = (_Float16)p1;
        }
#pragma unroll
        for (int t = 0; t < 8; ++t)
#pragma unroll
            for (int r = 0; r < 8; ++r) acc[t][r] *= alpha[r];

        asm volatile("s_wait_dscnt 0" ::: "memory"); // same-wave LDS RAW

        const _Float16* pb = pw + li * 32 + (hi ? 8 : 0);
        const v16h ap = ld2(pb, pb + 16);            // P as WMMA A (16x32)
#pragma unroll
        for (int t = 0; t < 8; ++t) {
            const _Float16* vb = Vm + (size_t)(t * 16 + li) * S_LEN + t0 + (hi ? 8 : 0);
            acc[t] = wmma16(ap, ld2(vb, vb + 16), acc[t]);
        }
    }
#pragma unroll
    for (int t = 0; t < 8; ++t)
#pragma unroll
        for (int r = 0; r < 8; ++r) {
            const int m = r + hi * 8;
            ctx[(size_t)(q0 + m) * HID + h * HD + t * 16 + li] =
                (_Float16)(acc[t][r] / lrow[r]);
        }
}

// ---------------------------------------------------------------------------
extern "C" void kernel_launch(void* const* d_in, const int* in_sizes, int n_in,
                              void* d_out, int out_size, void* d_ws, size_t ws_size,
                              hipStream_t stream) {
    const float* x   = (const float*)d_in[0];
    const float* wq  = (const float*)d_in[1];
    const float* wk  = (const float*)d_in[2];
    const float* wv  = (const float*)d_in[3];
    const float* wo  = (const float*)d_in[4];
    const float* qnw = (const float*)d_in[5];
    const float* knw = (const float*)d_in[6];
    float* out = (float*)d_out;

    char* p = (char*)d_ws;
    auto take = [&](size_t bytes) { char* r = p; p += (bytes + 255) & ~(size_t)255; return r; };

    _Float16* xh  = (_Float16*)take((size_t)S_LEN * HID * 2);
    _Float16* wqh = (_Float16*)take((size_t)HID * HID * 2);
    _Float16* wkh = (_Float16*)take((size_t)NKV * HD * HID * 2);
    _Float16* wvh = (_Float16*)take((size_t)NKV * HD * HID * 2);
    _Float16* woh = (_Float16*)take((size_t)HID * HID * 2);
    float*    q32 = (float*)   take((size_t)S_LEN * HID * 4);
    float*    k32 = (float*)   take((size_t)S_LEN * NKV * HD * 4);
    float*    v32 = (float*)   take((size_t)S_LEN * NKV * HD * 4);
    _Float16* qhd = (_Float16*)take((size_t)NH  * S_LEN * HD * 2);
    _Float16* khd = (_Float16*)take((size_t)NKV * S_LEN * HD * 2);
    _Float16* vTd = (_Float16*)take((size_t)NKV * HD * S_LEN * 2);
    _Float16* ctx = (_Float16*)q32;   // q32 is dead by the time attn writes ctx

    // 1) down-convert activations + weights to f16
    {
        int n;
        n = S_LEN * HID;      cvt_f16<<<(n + 1023) / 1024, 256, 0, stream>>>(x,  xh,  n);
        n = HID * HID;        cvt_f16<<<(n + 1023) / 1024, 256, 0, stream>>>(wq, wqh, n);
        n = NKV * HD * HID;   cvt_f16<<<(n + 1023) / 1024, 256, 0, stream>>>(wk, wkh, n);
        n = NKV * HD * HID;   cvt_f16<<<(n + 1023) / 1024, 256, 0, stream>>>(wv, wvh, n);
        n = HID * HID;        cvt_f16<<<(n + 1023) / 1024, 256, 0, stream>>>(wo, woh, n);
    }

    // 2) QKV projections (WMMA)
    {
        int waves, M = S_LEN;
        waves = (M >> 5) * (HID >> 6);
        gemm_wmma<<<waves / 4, 128, 0, stream>>>(xh, wqh, q32, M, HID, HID);
        waves = (M >> 5) * ((NKV * HD) >> 6);
        gemm_wmma<<<waves / 4, 128, 0, stream>>>(xh, wkh, k32, M, NKV * HD, HID);
        gemm_wmma<<<waves / 4, 128, 0, stream>>>(xh, wvh, v32, M, NKV * HD, HID);
    }

    // 3) per-head RMS norm + RoPE -> head-major f16; V transpose
    norm_rope<<<(NH  * S_LEN) / 8, 256, 0, stream>>>(q32, qnw, qhd, NH);
    norm_rope<<<(NKV * S_LEN) / 8, 256, 0, stream>>>(k32, knw, khd, NKV);
    {
        int n = NKV * HD * S_LEN;
        vtrans<<<(n + 1023) / 1024, 256, 0, stream>>>(v32, vTd);
    }

    // 4) flash causal GQA attention (WMMA for QK^T and PV)
    attn<<<(NH * (S_LEN / 16)) / 4, 128, 0, stream>>>(qhd, khd, vTd, ctx);

    // 5) output projection -> fp32 d_out
    {
        int waves = (S_LEN >> 5) * (HID >> 6);
        gemm_wmma<<<waves / 4, 128, 0, stream>>>(ctx, woh, out, S_LEN, HID, HID);
    }
}
